// NASBenchController_61838939128089
// MI455X (gfx1250) — matrix operand order repaired and checked by reading, not compiled, not run on hardware
//
#include <hip/hip_runtime.h>
#include <stdint.h>

// ---------------------------------------------------------------------------
// NASBench controller on MI455X (gfx1250, wave32).
// 18 fixed steps; per step: (1) wmma bf16 GEMV for LSTM gates (8192x2048 x2),
// (2) elementwise LSTM cell, (3) wmma GEMV hq = W_hid_attn @ h, (4) one-block
// sample/control kernel.
// If the workspace is large enough, the big weight matrices are converted to
// bf16 ONCE per launch (75.5 MB, L2-resident on the 192 MB L2) and the GEMV
// loop streams bf16 straight into v_wmma_f32_16x16x32_bf16 (halves the
// per-step weight traffic vs f32). Otherwise fall back to streaming f32 and
// converting in-register.
// ---------------------------------------------------------------------------

#define HDIM 2048
#define GDIM 8192
#define NSTEP 18

typedef __attribute__((ext_vector_type(16))) __bf16 v16bf;
typedef __attribute__((ext_vector_type(8)))  __bf16 v8bf;
typedef __attribute__((ext_vector_type(4)))  __bf16 v4bf;
typedef __attribute__((ext_vector_type(8)))  float  v8f;

// ---- workspace layout ------------------------------------------------------
// float region (offsets in floats)
#define OFF_X   0
#define OFF_H   (OFF_X + HDIM)
#define OFF_C   (OFF_H + HDIM)
#define OFF_G   (OFF_C + HDIM)
#define OFF_HQ  (OFF_G + GDIM)
#define OFF_E   (OFF_HQ + HDIM)
#define OFF_STF (OFF_E + 7 * HDIM)   // [0]=logp_sum [1]=ent_sum
#define OFF_STI (OFF_STF + 4)        // int: [0]=n
// bf16 region starts at byte offset BF_BASE (keep it 128B aligned)
#define BF_BASE   (131072)                       // bytes
#define NWIH      ((size_t)GDIM * HDIM)          // 16.7M elems
#define NWHH      ((size_t)GDIM * HDIM)
#define NWHID     ((size_t)HDIM * HDIM)
#define BF_WIH    0
#define BF_WHH    (BF_WIH + NWIH)
#define BF_WHID   (BF_WHH + NWHH)
#define BF_XV     (BF_WHID + NWHID)
#define BF_HV     (BF_XV + HDIM)
#define BF_END    (BF_HV + HDIM)
#define WS_NEED   (BF_BASE + BF_END * 2)         // bytes

// ---- tile builders ---------------------------------------------------------
// A-matrix 16x32 bf16 layout (wave32): lanes 0-15 hold K=0..7 (vgpr0-3) and
// K=16..23 (vgpr4-7); lanes 16-31 hold K=8..15 and K=24..31.  klo in {0,8}.
__device__ inline v16bf load_A16(const float* __restrict__ p, int klo) {
  float4 a0 = *reinterpret_cast<const float4*>(p + klo);
  float4 a1 = *reinterpret_cast<const float4*>(p + klo + 4);
  float4 a2 = *reinterpret_cast<const float4*>(p + klo + 16);
  float4 a3 = *reinterpret_cast<const float4*>(p + klo + 20);
  v16bf v;
  v[0]  = (__bf16)a0.x; v[1]  = (__bf16)a0.y; v[2]  = (__bf16)a0.z; v[3]  = (__bf16)a0.w;
  v[4]  = (__bf16)a1.x; v[5]  = (__bf16)a1.y; v[6]  = (__bf16)a1.z; v[7]  = (__bf16)a1.w;
  v[8]  = (__bf16)a2.x; v[9]  = (__bf16)a2.y; v[10] = (__bf16)a2.z; v[11] = (__bf16)a2.w;
  v[12] = (__bf16)a3.x; v[13] = (__bf16)a3.y; v[14] = (__bf16)a3.z; v[15] = (__bf16)a3.w;
  return v;
}
__device__ inline v16bf load_A16(const __bf16* __restrict__ p, int klo) {
  v8bf lo = *reinterpret_cast<const v8bf*>(p + klo);        // 16B load
  v8bf hi = *reinterpret_cast<const v8bf*>(p + klo + 16);   // 16B load
  return __builtin_shufflevector(lo, hi, 0, 1, 2, 3, 4, 5, 6, 7,
                                         8, 9, 10, 11, 12, 13, 14, 15);
}

// B-matrix 32x16 bf16: lanes 0-15 hold K=0..7,16..23, lanes 16-31 hold K=8..15,24..31;
// all 16 columns identical (vector broadcast). p points at x[k0 + (hi?8:0)].
__device__ inline v16bf load_B16(const float* __restrict__ p) {
  float4 b0 = *reinterpret_cast<const float4*>(p);
  float4 b1 = *reinterpret_cast<const float4*>(p + 4);
  float4 b2 = *reinterpret_cast<const float4*>(p + 16);
  float4 b3 = *reinterpret_cast<const float4*>(p + 20);
  v16bf v;
  v[0]  = (__bf16)b0.x; v[1]  = (__bf16)b0.y; v[2]  = (__bf16)b0.z; v[3]  = (__bf16)b0.w;
  v[4]  = (__bf16)b1.x; v[5]  = (__bf16)b1.y; v[6]  = (__bf16)b1.z; v[7]  = (__bf16)b1.w;
  v[8]  = (__bf16)b2.x; v[9]  = (__bf16)b2.y; v[10] = (__bf16)b2.z; v[11] = (__bf16)b2.w;
  v[12] = (__bf16)b3.x; v[13] = (__bf16)b3.y; v[14] = (__bf16)b3.z; v[15] = (__bf16)b3.w;
  return v;
}
__device__ inline v16bf load_B16(const __bf16* __restrict__ p) {
  v8bf lo = *reinterpret_cast<const v8bf*>(p);
  v8bf hi = *reinterpret_cast<const v8bf*>(p + 16);
  return __builtin_shufflevector(lo, hi, 0, 1, 2, 3, 4, 5, 6, 7,
                                         8, 9, 10, 11, 12, 13, 14, 15);
}

// ---- dual GEMV: out = W0@v0 + W1@v1 + b0 + b1 ------------------------------
template <typename WT>
__global__ __launch_bounds__(256) void wmma_gemv_dual_t(
    const WT* __restrict__ W0, const WT* __restrict__ W1,
    const WT* __restrict__ v0, const WT* __restrict__ v1,
    const float* __restrict__ b0, const float* __restrict__ b1,
    float* __restrict__ out) {
  int wave = blockIdx.x * (blockDim.x >> 5) + (threadIdx.x >> 5);
  int lane = threadIdx.x & 31;
  int row0 = wave << 4;
  int m    = lane & 15;
  bool hi  = lane >= 16;
  int klo  = hi ? 8 : 0;
  int kb   = hi ? 8 : 0;
  const WT* r0 = W0 + (size_t)(row0 + m) * HDIM;
  const WT* r1 = W1 + (size_t)(row0 + m) * HDIM;
  v8f acc = {};
  for (int k0 = 0; k0 < HDIM; k0 += 32) {
    __builtin_prefetch(r0 + k0 + 32, 0, 1);
    __builtin_prefetch(r1 + k0 + 32, 0, 1);
    v16bf aw = load_A16(r0 + k0, klo);
    v16bf bx = load_B16(v0 + k0 + kb);
    acc = __builtin_amdgcn_wmma_f32_16x16x32_bf16(false, aw, false, bx,
                                                  (short)0, acc, false, false);
    v16bf ah = load_A16(r1 + k0, klo);
    v16bf bh = load_B16(v1 + k0 + kb);
    acc = __builtin_amdgcn_wmma_f32_16x16x32_bf16(false, ah, false, bh,
                                                  (short)0, acc, false, false);
  }
  // D column N=0: lane0 holds M=0..7 (vgpr j), lane16 holds M=8..15.
  if (lane == 0) {
    for (int j = 0; j < 8; ++j) { int r = row0 + j;     out[r] = acc[j] + b0[r] + b1[r]; }
  } else if (lane == 16) {
    for (int j = 0; j < 8; ++j) { int r = row0 + 8 + j; out[r] = acc[j] + b0[r] + b1[r]; }
  }
}

// ---- single GEMV: out = W@v ------------------------------------------------
template <typename WT>
__global__ __launch_bounds__(256) void wmma_gemv_single_t(
    const WT* __restrict__ W, const WT* __restrict__ v,
    float* __restrict__ out) {
  int wave = blockIdx.x * (blockDim.x >> 5) + (threadIdx.x >> 5);
  int lane = threadIdx.x & 31;
  int row0 = wave << 4;
  int m    = lane & 15;
  bool hi  = lane >= 16;
  int klo  = hi ? 8 : 0;
  int kb   = hi ? 8 : 0;
  const WT* r0 = W + (size_t)(row0 + m) * HDIM;
  v8f acc = {};
  for (int k0 = 0; k0 < HDIM; k0 += 32) {
    __builtin_prefetch(r0 + k0 + 32, 0, 1);
    v16bf aw = load_A16(r0 + k0, klo);
    v16bf bx = load_B16(v + k0 + kb);
    acc = __builtin_amdgcn_wmma_f32_16x16x32_bf16(false, aw, false, bx,
                                                  (short)0, acc, false, false);
  }
  if (lane == 0) {
    for (int j = 0; j < 8; ++j) out[row0 + j] = acc[j];
  } else if (lane == 16) {
    for (int j = 0; j < 8; ++j) out[row0 + 8 + j] = acc[j];
  }
}

// ---- f32 -> bf16 weight mirror (run once per launch) -----------------------
__global__ __launch_bounds__(256) void cvt_bf16_kernel(
    const float* __restrict__ src, __bf16* __restrict__ dst, int n) {
  int stride = gridDim.x * blockDim.x * 4;
  for (int i = (blockIdx.x * blockDim.x + threadIdx.x) * 4; i < n; i += stride) {
    float4 f = *reinterpret_cast<const float4*>(src + i);
    v4bf o;
    o[0] = (__bf16)f.x; o[1] = (__bf16)f.y; o[2] = (__bf16)f.z; o[3] = (__bf16)f.w;
    *reinterpret_cast<v4bf*>(dst + i) = o;
  }
}

// ---- LSTM cell elementwise (optionally mirrors h into bf16) ----------------
__global__ __launch_bounds__(256) void lstm_gate_kernel(
    const float* __restrict__ g, float* __restrict__ h, float* __restrict__ c,
    __bf16* __restrict__ h_bf) {
  int i = blockIdx.x * blockDim.x + threadIdx.x;   // 0..2047
  float gi = g[i], gf = g[HDIM + i], gg = g[2 * HDIM + i], go = g[3 * HDIM + i];
  float si = 1.f / (1.f + expf(-gi));
  float sf = 1.f / (1.f + expf(-gf));
  float so = 1.f / (1.f + expf(-go));
  float c2 = sf * c[i] + si * tanhf(gg);
  float h2 = so * tanhf(c2);
  c[i] = c2;
  h[i] = h2;
  if (h_bf) h_bf[i] = (__bf16)h2;
}

// ---- E = node_emb @ w_emb_attn^T  (7 x 2048), run once ---------------------
__global__ __launch_bounds__(256) void emb_attn_kernel(
    const float* __restrict__ node_emb, const float* __restrict__ w_emb_attn,
    float* __restrict__ E) {
  int wave = blockIdx.x * (blockDim.x >> 5) + (threadIdx.x >> 5);
  int lane = threadIdx.x & 31;
  if (wave >= 7 * HDIM) return;
  int r = wave >> 11;          // node_emb row
  int cIdx = wave & (HDIM - 1);
  const float* wr = w_emb_attn + (size_t)cIdx * HDIM;
  const float* er = node_emb + (size_t)r * HDIM;
  float s = 0.f;
  for (int k = lane; k < HDIM; k += 32) s += er[k] * wr[k];
  for (int off = 16; off; off >>= 1) s += __shfl_down(s, off, 32);
  if (lane == 0) E[wave] = s;
}

// ---- init x/h/c/state (+ optional bf16 mirrors) ----------------------------
__global__ __launch_bounds__(256) void init_state_kernel(
    float* __restrict__ ws, __bf16* __restrict__ x_bf, __bf16* __restrict__ h_bf) {
  int i = blockIdx.x * blockDim.x + threadIdx.x;
  if (i < 3 * HDIM) ws[OFF_X + i] = 0.f;
  if (x_bf && i < HDIM) { x_bf[i] = (__bf16)0.f; h_bf[i] = (__bf16)0.f; }
  if (i == 0) {
    ws[OFF_STF] = 0.f;
    ws[OFF_STF + 1] = 0.f;
    reinterpret_cast<int*>(ws + OFF_STI)[0] = 0;
  }
}

// ---- Threefry-2x32 (20 rounds), JAX-style ----------------------------------
__device__ inline uint2 threefry2x32(unsigned k0, unsigned k1,
                                     unsigned c0, unsigned c1) {
  const unsigned R[8] = {13, 15, 26, 6, 17, 29, 16, 24};
  unsigned ks[3] = {k0, k1, k0 ^ k1 ^ 0x1BD11BDAu};
  unsigned x0 = c0 + k0, x1 = c1 + k1;
  for (int i = 0; i < 5; ++i) {
    for (int r = 0; r < 4; ++r) {
      x0 += x1;
      unsigned rot = R[(i & 1) * 4 + r];
      x1 = (x1 << rot) | (x1 >> (32 - rot));
      x1 ^= x0;
    }
    x0 += ks[(i + 1) % 3];
    x1 += ks[(i + 2) % 3] + (unsigned)(i + 1);
  }
  return make_uint2(x0, x1);
}

// ---- one control step: logits -> sample -> update x ------------------------
__global__ __launch_bounds__(256) void control_step_kernel(
    int t,
    const float* __restrict__ n_nodes_emb, const float* __restrict__ node_emb,
    const float* __restrict__ op_emb, const float* __restrict__ w_v,
    const float* __restrict__ w_soft_w, const float* __restrict__ w_soft_b,
    const float* __restrict__ n_soft_w, const float* __restrict__ n_soft_b,
    const float* __restrict__ h, const float* __restrict__ hq,
    const float* __restrict__ E,
    float* __restrict__ x, __bf16* __restrict__ x_bf,
    float* __restrict__ stf, int* __restrict__ sti,
    int* __restrict__ arch_out, float* __restrict__ scal_out) {
  __shared__ float partial[8];
  __shared__ float logits_s[8];
  __shared__ int a_sh;
  int tid = threadIdx.x, lane = tid & 31, wv = tid >> 5;

  int n = (t == 0) ? 0 : sti[0];
  int type, rows;                       // 0: node-count head, 1: attention, 2: op head
  if (t == 0)               { type = 0; rows = 5; }
  else if (t <= 2 * n + 1)  { if (t & 1) { type = 1; rows = ((t - 1) >> 1) + 1; }
                              else       { type = 2; rows = 3; } }
  else                      { type = 1; rows = n + 2; }

  for (int r = 0; r < rows; ++r) {
    float s = 0.f;
    if (type == 1) {
      const float* Er = E + (size_t)r * HDIM;
      for (int k = tid; k < HDIM; k += 256) {
        float q = tanhf(Er[k] + hq[k]);
        s += q * w_v[k];
      }
    } else {
      const float* wr = (type == 0 ? n_soft_w : w_soft_w) + (size_t)r * HDIM;
      for (int k = tid; k < HDIM; k += 256) s += h[k] * wr[k];
    }
    for (int off = 16; off; off >>= 1) s += __shfl_down(s, off, 32);
    if (lane == 0) partial[wv] = s;
    __syncthreads();
    if (tid == 0) {
      float tot = 0.f;
      for (int w = 0; w < 8; ++w) tot += partial[w];
      logits_s[r] = tot;
    }
    __syncthreads();
  }

  if (tid == 0) {
    float lg[8];
    for (int r = 0; r < rows; ++r) {
      float L = logits_s[r];
      if (type == 1) L = 1.5f * tanhf(L);
      else {
        const float* b = (type == 0 ? n_soft_b : w_soft_b);
        L = (1.5f / 2.5f) * tanhf(L + b[r]);
      }
      lg[r] = L;
    }
    float mx = lg[0];
    for (int r = 1; r < rows; ++r) mx = fmaxf(mx, lg[r]);
    float se = 0.f;
    for (int r = 0; r < rows; ++r) se += expf(lg[r] - mx);
    float lse = logf(se) + mx;

    // key = fold_in(key(42), t); Gumbel-argmax categorical
    uint2 key = threefry2x32(0u, 42u, 0u, (unsigned)t);
    int half = (rows + 1) >> 1;
    float best = -1e30f; int a = 0;
    for (int i = 0; i < half; ++i) {
      uint2 bits = threefry2x32(key.x, key.y, (unsigned)i, (unsigned)(half + i));
      unsigned bs[2] = {bits.x, bits.y};
      for (int j = 0; j < 2; ++j) {
        int r = i + j * half;
        if (r < rows) {
          float u = ((bs[j] >> 9) + 0.5f) * (1.f / 8388608.f);
          float gmb = -logf(-logf(u));
          float val = lg[r] + gmb;
          if (val > best) { best = val; a = r; }
        }
      }
    }
    float lp_a = lg[a] - lse;
    float ent = 0.f;
    for (int r = 0; r < rows; ++r) {
      float lp = lg[r] - lse;
      ent -= expf(lp) * lp;
    }
    float lacc = (t == 0) ? 0.f : stf[0];
    float eacc = (t == 0) ? 0.f : stf[1];
    lacc += lp_a; eacc += ent;
    stf[0] = lacc; stf[1] = eacc;
    if (t == 0) { sti[0] = a + 1; arch_out[0] = a + 1; }
    else arch_out[t] = a;
    if (t == NSTEP - 1) { scal_out[0] = lacc; scal_out[1] = eacc; }
    a_sh = a;
  }
  __syncthreads();
  int a = a_sh;
  const float* src = (t == 0) ? (n_nodes_emb + (size_t)a * HDIM)
                   : (type == 1 ? node_emb + (size_t)a * HDIM
                                : op_emb + (size_t)a * HDIM);
  for (int k = tid; k < HDIM; k += 256) {
    float v = src[k];
    x[k] = v;
    if (x_bf) x_bf[k] = (__bf16)v;
  }
}

// ---------------------------------------------------------------------------
extern "C" void kernel_launch(void* const* d_in, const int* in_sizes, int n_in,
                              void* d_out, int out_size, void* d_ws, size_t ws_size,
                              hipStream_t stream) {
  const float* n_nodes_emb = (const float*)d_in[0];
  const float* node_emb    = (const float*)d_in[1];
  const float* op_emb      = (const float*)d_in[2];
  const float* w_ih        = (const float*)d_in[3];
  const float* w_hh        = (const float*)d_in[4];
  const float* b_ih        = (const float*)d_in[5];
  const float* b_hh        = (const float*)d_in[6];
  const float* w_emb_attn  = (const float*)d_in[7];
  const float* w_hid_attn  = (const float*)d_in[8];
  const float* w_v         = (const float*)d_in[9];
  const float* w_soft_w    = (const float*)d_in[10];
  const float* w_soft_b    = (const float*)d_in[11];
  const float* n_soft_w    = (const float*)d_in[12];
  const float* n_soft_b    = (const float*)d_in[13];

  float* ws  = (float*)d_ws;
  float* x   = ws + OFF_X;
  float* h   = ws + OFF_H;
  float* c   = ws + OFF_C;
  float* g   = ws + OFF_G;
  float* hq  = ws + OFF_HQ;
  float* E   = ws + OFF_E;
  float* stf = ws + OFF_STF;
  int*   sti = (int*)(ws + OFF_STI);

  int*   arch = (int*)d_out;              // arch_seq[18] (int32)
  float* scal = (float*)d_out + NSTEP;    // logp_sum, ent_sum

  const bool use_bf = ws_size >= (size_t)WS_NEED;   // fixed per process: deterministic
  __bf16* bf      = (__bf16*)((char*)d_ws + BF_BASE);
  __bf16* wih_bf  = use_bf ? bf + BF_WIH  : nullptr;
  __bf16* whh_bf  = use_bf ? bf + BF_WHH  : nullptr;
  __bf16* whid_bf = use_bf ? bf + BF_WHID : nullptr;
  __bf16* x_bf    = use_bf ? bf + BF_XV   : nullptr;
  __bf16* h_bf    = use_bf ? bf + BF_HV   : nullptr;

  init_state_kernel<<<24, 256, 0, stream>>>(ws, x_bf, h_bf);
  emb_attn_kernel<<<(7 * HDIM) / 8, 256, 0, stream>>>(node_emb, w_emb_attn, E);

  if (use_bf) {
    cvt_bf16_kernel<<<4096, 256, 0, stream>>>(w_ih, wih_bf, (int)NWIH);
    cvt_bf16_kernel<<<4096, 256, 0, stream>>>(w_hh, whh_bf, (int)NWHH);
    cvt_bf16_kernel<<<1024, 256, 0, stream>>>(w_hid_attn, whid_bf, (int)NWHID);
  }

  for (int t = 0; t < NSTEP; ++t) {
    if (use_bf) {
      wmma_gemv_dual_t<__bf16><<<64, 256, 0, stream>>>(
          wih_bf, whh_bf, x_bf, h_bf, b_ih, b_hh, g);
    } else {
      wmma_gemv_dual_t<float><<<64, 256, 0, stream>>>(
          w_ih, w_hh, x, h, b_ih, b_hh, g);
    }
    lstm_gate_kernel<<<HDIM / 256, 256, 0, stream>>>(g, h, c, h_bf);
    if (use_bf) {
      wmma_gemv_single_t<__bf16><<<16, 256, 0, stream>>>(whid_bf, h_bf, hq);
    } else {
      wmma_gemv_single_t<float><<<16, 256, 0, stream>>>(w_hid_attn, h, hq);
    }
    control_step_kernel<<<1, 256, 0, stream>>>(
        t, n_nodes_emb, node_emb, op_emb, w_v, w_soft_w, w_soft_b,
        n_soft_w, n_soft_b, h, hq, E, x, x_bf, stf, sti, arch, scal);
  }
}
